// Qwen3_5FullAttention_31791347925072
// MI455X (gfx1250) — compile-verified, matrix operand order checked by reading, not run on hardware
//
#include <hip/hip_runtime.h>
#include <hip/hip_bf16.h>

#ifndef __has_builtin
#define __has_builtin(x) 0
#endif

typedef unsigned int   uint32;
typedef unsigned short ushort16;

typedef __attribute__((ext_vector_type(16))) __bf16 v16bf;
typedef __attribute__((ext_vector_type(8)))  float  v8f;
typedef __attribute__((ext_vector_type(4)))  uint32 v4u;
typedef __attribute__((ext_vector_type(4)))  int    v4i;

union Frag {
    v16bf bf;
    v4u   u[2];
};

#define S_SEQ 2048
#define HID   4096
#define NH    32
#define NKV   8
#define HD    128
#define QSCALE 0.08838834764831845f   // 128^-0.5

// LDS staging geometry for the GEMM: 32-wide K slab, rows padded to 40 halves
#define LROW   40
#define ABUF_H (128 * LROW)           // 5120 halves
#define BUF_H  (2 * ABUF_H)           // A + B regions per buffer

__device__ __forceinline__ ushort16 f2bf(float f) {
    uint32 u = __builtin_bit_cast(uint32, f);
    u += 0x7FFFu + ((u >> 16) & 1u);       // round-to-nearest-even
    return (ushort16)(u >> 16);
}

__device__ __forceinline__ v8f wmma_bf16(const Frag& a, const Frag& b, v8f c) {
    return __builtin_amdgcn_wmma_f32_16x16x32_bf16(
        false, a.bf, false, b.bf, (short)0, c, false, false);
}

// ---- async Global -> LDS 16-byte copy (gfx1250 path) -----------------------
#if defined(__gfx1250__)
#define AS1 __attribute__((address_space(1)))
#define AS3 __attribute__((address_space(3)))
#if __has_builtin(__builtin_amdgcn_global_load_async_to_lds_b128)
#define ASYNC_VIA_BUILTIN 1
#endif
#endif

__device__ __forceinline__ void cp_g2l_b128(const ushort16* g, ushort16* l) {
#if defined(ASYNC_VIA_BUILTIN)
    __builtin_amdgcn_global_load_async_to_lds_b128(
        (AS1 v4i*)(void*)g, (AS3 v4i*)(void*)l, 0, 0);
#elif defined(__gfx1250__)
    asm volatile("global_load_async_to_lds_b128 %0, %1, off"
                 :: "v"((uint32)(uintptr_t)l), "v"((const void*)g)
                 : "memory");
#else
    *(v4u*)l = *(const v4u*)g;      // host-parse fallback (never codegen'd)
#endif
}

__device__ __forceinline__ void wait_async0() {
#if defined(__gfx1250__)
#if __has_builtin(__builtin_amdgcn_s_wait_asynccnt)
    __builtin_amdgcn_s_wait_asynccnt(0);
#else
    asm volatile("s_wait_asynccnt 0" ::: "memory");
#endif
#endif
}

// ---------------- elementwise fp32 -> bf16 ----------------
__global__ void k_cvt_bf16(const float* __restrict__ in, ushort16* __restrict__ out, size_t n) {
    size_t i = (size_t)blockIdx.x * blockDim.x + threadIdx.x;
    if (i < n) out[i] = f2bf(in[i]);
}

// ------- LDS-tiled transpose-convert: fp32 [K][N] -> bf16 [N][K] -------
// block (32,8); K,N multiples of 32
__global__ void k_cvt_transpose(const float* __restrict__ in, ushort16* __restrict__ out,
                                int K, int N) {
    __shared__ ushort16 tile[32][33];
    const int kt = blockIdx.y * 32;
    const int nt = blockIdx.x * 32;
    const int tx = threadIdx.x, ty = threadIdx.y;
#pragma unroll
    for (int i = 0; i < 4; ++i) {
        int k = kt + ty + i * 8;
        tile[ty + i * 8][tx] = f2bf(in[(size_t)k * N + nt + tx]);
    }
    __syncthreads();
#pragma unroll
    for (int i = 0; i < 4; ++i) {
        int n = nt + ty + i * 8;
        out[(size_t)n * K + kt + tx] = tile[tx][ty + i * 8];
    }
}

// ---------------- bf16 GEMM: C[M,N] = A[M,K] * Bt[N,K]^T ----------------
// block = 256 threads (8 waves, 4 in M x 2 in N); wave tile 32x64; block tile 128x128
// double-buffered async LDS staging of 128x32 A and 128x32 B slabs
__device__ __forceinline__ void stage_tile(ushort16* sb, const ushort16* A, const ushort16* Bt,
                                           int mblk, int nblk, int K, int kc, int tid) {
#pragma unroll
    for (int j = 0; j < 2; ++j) {
        int idx = tid + j * 256;
        int r = idx >> 2;
        int c = (idx & 3) * 8;
        cp_g2l_b128(A  + (size_t)(mblk + r) * K + kc + c, sb + r * LROW + c);
        cp_g2l_b128(Bt + (size_t)(nblk + r) * K + kc + c, sb + ABUF_H + r * LROW + c);
    }
}

__global__ void k_gemm_bf16(const ushort16* __restrict__ A, const ushort16* __restrict__ Bt,
                            float* __restrict__ C, int M, int N, int K) {
    __shared__ ushort16 sbuf[2 * BUF_H];    // 40 KB
    const int tid   = threadIdx.x;
    const int lane  = tid & 31;
    const int wv    = tid >> 5;
    const int wm    = wv & 3;               // wave row group (4)
    const int wn    = wv >> 2;              // wave col group (2)
    const int mblk  = blockIdx.y * 128;
    const int nblk  = blockIdx.x * 128;
    const int mr    = lane & 15;
    const int kbase = (lane >> 4) * 8;

    v8f acc[2][4] = {};

    stage_tile(sbuf, A, Bt, mblk, nblk, K, 0, tid);
    wait_async0();
    __syncthreads();

    for (int kc = 0; kc < K; kc += 32) {
        const int p = (kc >> 5) & 1;
        ushort16* sb = sbuf + p * BUF_H;
        if (kc + 32 < K)
            stage_tile(sbuf + (p ^ 1) * BUF_H, A, Bt, mblk, nblk, K, kc + 32, tid);

        Frag af[2];
#pragma unroll
        for (int f = 0; f < 2; ++f) {
            const ushort16* ar = sb + (wm * 32 + f * 16 + mr) * LROW;
            af[f].u[0] = *(const v4u*)(ar + kbase);
            af[f].u[1] = *(const v4u*)(ar + 16 + kbase);
        }
#pragma unroll
        for (int t = 0; t < 4; ++t) {
            const ushort16* br = sb + ABUF_H + (wn * 64 + t * 16 + mr) * LROW;
            Frag bf;
            bf.u[0] = *(const v4u*)(br + kbase);
            bf.u[1] = *(const v4u*)(br + 16 + kbase);
#pragma unroll
            for (int f = 0; f < 2; ++f)
                acc[f][t] = wmma_bf16(af[f], bf, acc[f][t]);
        }
        wait_async0();
        __syncthreads();
    }

    const int rbase = kbase;                // (lane>>4)*8
#pragma unroll
    for (int f = 0; f < 2; ++f)
#pragma unroll
        for (int t = 0; t < 4; ++t)
#pragma unroll
            for (int r = 0; r < 8; ++r)
                C[(size_t)(mblk + wm * 32 + f * 16 + rbase + r) * N
                  + nblk + wn * 64 + t * 16 + mr] = acc[f][t][r];
}

// ---------------- Q post: RMS-norm + partial RoPE -> bf16 [S][H][D] ----------------
__global__ void k_q_post(const float* __restrict__ qg, const int* __restrict__ pos,
                         const float* __restrict__ nw, ushort16* __restrict__ qb) {
    const int lane = threadIdx.x & 31;
    const int gw   = blockIdx.x * 8 + (threadIdx.x >> 5);   // over S*H
    const int s    = gw >> 5;
    const int h    = gw & 31;
    const float* x = qg + ((size_t)s * NH + h) * (2 * HD);  // q = first 128 of 256
    float v[4], ss = 0.f;
#pragma unroll
    for (int i = 0; i < 4; ++i) { v[i] = x[lane + 32 * i]; ss += v[i] * v[i]; }
#pragma unroll
    for (int off = 16; off >= 1; off >>= 1) ss += __shfl_xor(ss, off, 32);
    float rms = rsqrtf(ss * (1.0f / 128.0f) + 1e-6f);
#pragma unroll
    for (int i = 0; i < 4; ++i) v[i] *= rms * (1.0f + nw[lane + 32 * i]);
    float p   = (float)pos[s];
    float ang = p * __powf(10000.0f, -(float)(lane & 15) / 16.0f);
    float cs = __cosf(ang), sn = __sinf(ang);
    float other = __shfl_xor(v[0], 16, 32);
    v[0] = (lane < 16) ? (v[0] * cs - other * sn) : (v[0] * cs + other * sn);
    ushort16* out = qb + ((size_t)s * NH + h) * HD;
#pragma unroll
    for (int i = 0; i < 4; ++i) out[lane + 32 * i] = f2bf(v[i]);
}

// ---------------- K post: RMS-norm + partial RoPE -> bf16 [S][KV][D] ----------------
__global__ void k_k_post(const float* __restrict__ kf, const int* __restrict__ pos,
                         const float* __restrict__ nw, ushort16* __restrict__ kb) {
    const int lane = threadIdx.x & 31;
    const int gw   = blockIdx.x * 8 + (threadIdx.x >> 5);   // over S*KV
    const int s    = gw >> 3;
    const int kv   = gw & 7;
    const float* x = kf + (size_t)s * (NKV * HD) + kv * HD;
    float v[4], ss = 0.f;
#pragma unroll
    for (int i = 0; i < 4; ++i) { v[i] = x[lane + 32 * i]; ss += v[i] * v[i]; }
#pragma unroll
    for (int off = 16; off >= 1; off >>= 1) ss += __shfl_xor(ss, off, 32);
    float rms = rsqrtf(ss * (1.0f / 128.0f) + 1e-6f);
#pragma unroll
    for (int i = 0; i < 4; ++i) v[i] *= rms * (1.0f + nw[lane + 32 * i]);
    float p   = (float)pos[s];
    float ang = p * __powf(10000.0f, -(float)(lane & 15) / 16.0f);
    float cs = __cosf(ang), sn = __sinf(ang);
    float other = __shfl_xor(v[0], 16, 32);
    v[0] = (lane < 16) ? (v[0] * cs - other * sn) : (v[0] * cs + other * sn);
    ushort16* out = kb + ((size_t)s * NKV + kv) * HD;
#pragma unroll
    for (int i = 0; i < 4; ++i) out[lane + 32 * i] = f2bf(v[i]);
}

// ---------------- Flash attention (causal, GQA) ----------------
// one wave per (head, 16-query tile); key tiles of 32
__global__ void k_flash(const ushort16* __restrict__ Qb, const ushort16* __restrict__ Kb,
                        const ushort16* __restrict__ Vt, float* __restrict__ O) {
    __shared__ ushort16 plds[8 * 512];    // per-wave 16x32 bf16 P staging
    const int lane  = threadIdx.x & 31;
    const int wv    = threadIdx.x >> 5;
    const int gw    = blockIdx.x * 8 + wv;      // over NH * (S/16)
    const int h     = gw >> 7;
    const int qt    = gw & 127;
    const int q0    = qt * 16;
    const int kvh   = h >> 2;                   // group = 4
    const int mr    = lane & 15;
    const int kbase = (lane >> 4) * 8;
    const int rbase = kbase;

    Frag qf[4];
    const ushort16* qrow = Qb + ((size_t)(q0 + mr) * NH + h) * HD;
#pragma unroll
    for (int c = 0; c < 4; ++c) {
        qf[c].u[0] = *(const v4u*)(qrow + c * 32 + kbase);
        qf[c].u[1] = *(const v4u*)(qrow + c * 32 + 16 + kbase);
    }

    v8f o[8] = {};
    float mrun[8], lrun[8];
#pragma unroll
    for (int r = 0; r < 8; ++r) { mrun[r] = -1e30f; lrun[r] = 0.f; }

    const int nkt = (q0 + 15) / 32 + 1;
    for (int kt = 0; kt < nkt; ++kt) {
        const int k0 = kt * 32;
        v8f s0 = {}, s1 = {};
#pragma unroll
        for (int c = 0; c < 4; ++c) {
            const ushort16* kr0 = Kb + ((size_t)(k0 + mr) * NKV + kvh) * HD + c * 32;
            Frag b0;
            b0.u[0] = *(const v4u*)(kr0 + kbase);
            b0.u[1] = *(const v4u*)(kr0 + 16 + kbase);
            s0 = wmma_bf16(qf[c], b0, s0);
            const ushort16* kr1 = Kb + ((size_t)(k0 + 16 + mr) * NKV + kvh) * HD + c * 32;
            Frag b1;
            b1.u[0] = *(const v4u*)(kr1 + kbase);
            b1.u[1] = *(const v4u*)(kr1 + 16 + kbase);
            s1 = wmma_bf16(qf[c], b1, s1);
        }
        const bool need_mask = (k0 + 31) > (q0 + 15);
#pragma unroll
        for (int r = 0; r < 8; ++r) {
            float a0 = s0[r] * QSCALE;
            float a1 = s1[r] * QSCALE;
            if (need_mask) {
                int row = q0 + rbase + r;
                if (k0 + mr > row)      a0 = -1e30f;
                if (k0 + 16 + mr > row) a1 = -1e30f;
            }
            float vmax = fmaxf(a0, a1);
            vmax = fmaxf(vmax, __shfl_xor(vmax, 1, 32));
            vmax = fmaxf(vmax, __shfl_xor(vmax, 2, 32));
            vmax = fmaxf(vmax, __shfl_xor(vmax, 4, 32));
            vmax = fmaxf(vmax, __shfl_xor(vmax, 8, 32));
            float mnew  = fmaxf(mrun[r], vmax);
            float alpha = __expf(mrun[r] - mnew);
            mrun[r] = mnew;
            float p0 = __expf(a0 - mnew);
            float p1 = __expf(a1 - mnew);
            float rs = p0 + p1;
            rs += __shfl_xor(rs, 1, 32);
            rs += __shfl_xor(rs, 2, 32);
            rs += __shfl_xor(rs, 4, 32);
            rs += __shfl_xor(rs, 8, 32);
            lrun[r] = lrun[r] * alpha + rs;
#pragma unroll
            for (int a = 0; a < 8; ++a) o[a][r] *= alpha;
            int li = wv * 512 + (rbase + r) * 32;
            plds[li + mr]      = f2bf(p0);
            plds[li + 16 + mr] = f2bf(p1);
        }
        asm volatile("s_wait_dscnt 0" ::: "memory");
        Frag pf;
        const ushort16* pl = plds + wv * 512 + mr * 32;
        pf.u[0] = *(const v4u*)(pl + kbase);
        pf.u[1] = *(const v4u*)(pl + 16 + kbase);
#pragma unroll
        for (int n = 0; n < 8; ++n) {
            const ushort16* vr = Vt + ((size_t)kvh * HD + n * 16 + mr) * S_SEQ + k0;
            Frag vb;
            vb.u[0] = *(const v4u*)(vr + kbase);
            vb.u[1] = *(const v4u*)(vr + 16 + kbase);
            o[n] = wmma_bf16(pf, vb, o[n]);
        }
    }
#pragma unroll
    for (int n = 0; n < 8; ++n) {
#pragma unroll
        for (int r = 0; r < 8; ++r) {
            float val = o[n][r] / lrun[r];
            O[(size_t)(q0 + rbase + r) * (NH * HD) + h * HD + n * 16 + mr] = val;
        }
    }
}

// ---------------- gating: gated = bf16(attn * sigmoid(gate)) ----------------
__global__ void k_gate(const float* __restrict__ attn, const float* __restrict__ qg,
                       ushort16* __restrict__ gated) {
    size_t i = (size_t)blockIdx.x * blockDim.x + threadIdx.x;
    if (i >= (size_t)S_SEQ * NH * HD) return;
    int s = (int)(i >> 12);
    int j = (int)(i & 4095);
    int h = j >> 7, d = j & 127;
    float g = qg[(size_t)s * (NH * 2 * HD) + h * 256 + 128 + d];
    float sig = 1.0f / (1.0f + __expf(-g));
    gated[i] = f2bf(attn[i] * sig);
}

extern "C" void kernel_launch(void* const* d_in, const int* in_sizes, int n_in,
                              void* d_out, int out_size, void* d_ws, size_t ws_size,
                              hipStream_t stream) {
    const float* hidden = (const float*)d_in[0];
    const int*   pos    = (const int*)d_in[1];
    const float* Wq     = (const float*)d_in[2];
    const float* Wk     = (const float*)d_in[3];
    const float* Wv     = (const float*)d_in[4];
    const float* Wo     = (const float*)d_in[5];
    const float* qnw    = (const float*)d_in[6];
    const float* knw    = (const float*)d_in[7];
    float* out = (float*)d_out;

    char* p = (char*)d_ws;
    auto take = [&](size_t bytes) { char* q = p; p += bytes; return q; };
    ushort16* hsB   = (ushort16*)take((size_t)S_SEQ * HID * 2);
    ushort16* WqT   = (ushort16*)take((size_t)(2 * NH * HD) * HID * 2);
    ushort16* WkT   = (ushort16*)take((size_t)(NKV * HD) * HID * 2);
    ushort16* WvT   = (ushort16*)take((size_t)(NKV * HD) * HID * 2);
    ushort16* WoT   = (ushort16*)take((size_t)HID * (NH * HD) * 2);
    float*    qg    = (float*)take((size_t)S_SEQ * (2 * NH * HD) * 4);
    float*    kf    = (float*)take((size_t)S_SEQ * (NKV * HD) * 4);
    float*    vf    = (float*)take((size_t)S_SEQ * (NKV * HD) * 4);
    ushort16* qb    = (ushort16*)take((size_t)S_SEQ * NH * HD * 2);
    ushort16* kb    = (ushort16*)take((size_t)S_SEQ * NKV * HD * 2);
    ushort16* vt    = (ushort16*)take((size_t)S_SEQ * NKV * HD * 2);
    float*    attn  = (float*)take((size_t)S_SEQ * NH * HD * 4);
    ushort16* gated = (ushort16*)take((size_t)S_SEQ * NH * HD * 2);
    (void)ws_size; (void)n_in; (void)in_sizes; (void)out_size;

    const dim3 tb(32, 8);

    // 1) convert activations + tiled transpose-convert all weights to bf16
    {
        size_t n = (size_t)S_SEQ * HID;
        k_cvt_bf16<<<dim3((n + 255) / 256), dim3(256), 0, stream>>>(hidden, hsB, n);
    }
    k_cvt_transpose<<<dim3((2 * NH * HD) / 32, HID / 32), tb, 0, stream>>>(Wq, WqT, HID, 2 * NH * HD);
    k_cvt_transpose<<<dim3((NKV * HD) / 32, HID / 32), tb, 0, stream>>>(Wk, WkT, HID, NKV * HD);
    k_cvt_transpose<<<dim3((NKV * HD) / 32, HID / 32), tb, 0, stream>>>(Wv, WvT, HID, NKV * HD);
    k_cvt_transpose<<<dim3(HID / 32, (NH * HD) / 32), tb, 0, stream>>>(Wo, WoT, NH * HD, HID);

    // 2) projections (WMMA GEMMs, async-LDS double buffered)
    k_gemm_bf16<<<dim3((2 * NH * HD) / 128, S_SEQ / 128), dim3(256), 0, stream>>>(
        hsB, WqT, qg, S_SEQ, 2 * NH * HD, HID);
    k_gemm_bf16<<<dim3((NKV * HD) / 128, S_SEQ / 128), dim3(256), 0, stream>>>(
        hsB, WkT, kf, S_SEQ, NKV * HD, HID);
    k_gemm_bf16<<<dim3((NKV * HD) / 128, S_SEQ / 128), dim3(256), 0, stream>>>(
        hsB, WvT, vf, S_SEQ, NKV * HD, HID);

    // 3) norm + rope + V transpose (fp32 [S][KV*D] -> bf16 [KV*D][S])
    k_q_post<<<dim3((S_SEQ * NH) / 8), dim3(256), 0, stream>>>(qg, pos, qnw, qb);
    k_k_post<<<dim3((S_SEQ * NKV) / 8), dim3(256), 0, stream>>>(kf, pos, knw, kb);
    k_cvt_transpose<<<dim3((NKV * HD) / 32, S_SEQ / 32), tb, 0, stream>>>(vf, vt, S_SEQ, NKV * HD);

    // 4) flash attention (WMMA)
    k_flash<<<dim3((NH * (S_SEQ / 16)) / 8), dim3(256), 0, stream>>>(qb, kb, vt, attn);

    // 5) gating + output projection (WMMA)
    {
        size_t n = (size_t)S_SEQ * NH * HD;
        k_gate<<<dim3((n + 255) / 256), dim3(256), 0, stream>>>(attn, qg, gated);
    }
    k_gemm_bf16<<<dim3(HID / 128, S_SEQ / 128), dim3(256), 0, stream>>>(
        gated, WoT, out, S_SEQ, HID, NH * HD);
}
